// GNNDiffPool_807453851812
// MI455X (gfx1250) — matrix operand order repaired and checked by reading, not compiled, not run on hardware
//
#include <hip/hip_runtime.h>
#include <hip/hip_bf16.h>

// GNN DiffPool forward, algebraically simplified:
//   - Ap = S^T A S is dead code (never used in output) -> skip dense adjacency.
//   - Xp.mean(axis=1) = (1/K) * sum_n NE[n,:]  since softmax rows sum to 1
//     -> assignment GCN (W_asn) drops out entirely; pool = mean over nodes.
// Pipeline: dinv -> [x@W_pre (WMMA)] -> scatter (LDS ds_add) -> relu = h
//           -> [h@W_emb (WMMA)] -> scatter + fused mean-pool -> MLP head.

typedef __attribute__((ext_vector_type(16))) _Float16 v16h;
typedef __attribute__((ext_vector_type(8)))  float    v8f;
typedef __attribute__((ext_vector_type(4)))  float    f4;

static constexpr int Bg  = 128;     // graphs
static constexpr int NPG = 1000;    // nodes / graph
static constexpr int EPG = 16000;   // edges / graph
static constexpr int Dm  = 128;     // feature dim
static constexpr int Kc  = 64;      // clusters (only 1/K survives)
static constexpr int Cc  = 10;      // classes
static constexpr int Nn  = Bg * NPG;    // 128000
static constexpr int Ee  = Bg * EPG;    // 2048000

// ---------------------------------------------------------------- utilities
__global__ void zero_i32_kernel(int* __restrict__ p, int n) {
    int i = blockIdx.x * blockDim.x + threadIdx.x;
    if (i < n) p[i] = 0;
}

__global__ void deg_count_kernel(const int* __restrict__ dst, int e, int* __restrict__ deg) {
    int i = blockIdx.x * blockDim.x + threadIdx.x;
    if (i < e) atomicAdd(&deg[dst[i]], 1);
}

__global__ void make_dinv_kernel(const int* __restrict__ deg, float* __restrict__ dinv, int n) {
    int i = blockIdx.x * blockDim.x + threadIdx.x;
    if (i < n) dinv[i] = rsqrtf((float)(deg[i] + 1));   // +1: self loop
}

// ------------------------------------------------------- WMMA GEMM  Y = X@W
// X:[rows,128] f32, W:[128,128] f32, Y:[rows,128] f32.  f16 inputs, f32 acc.
// Block = 256 thr = 8 waves; wave w owns output columns [16w,16w+16).
// B fragments hoisted (loaded once); A tile staged via LDS (16x128 f32).
__global__ void __launch_bounds__(256)
gemm128_wmma_kernel(const float* __restrict__ X, const float* __restrict__ W,
                    float* __restrict__ Y, int nRowTiles, int tilesPerBlock) {
    extern __shared__ float sA[];                 // 16*128 floats = 8KB
    const int lane = threadIdx.x & 31;
    const int wave = threadIdx.x >> 5;
    const int half = lane >> 4;                   // 0/1
    const int ln   = lane & 15;                   // m for A, n for B/D
    const int ncol = wave * 16 + ln;

    // B fragment, 32x16 f16: lane holds column ncol, K = 32t + 16*half + j
    v16h bfrag[4];
#pragma unroll
    for (int t = 0; t < 4; ++t) {
        v16h b;
#pragma unroll
        for (int j = 0; j < 16; ++j) {
            int k = 32 * t + 16 * half + j;
            b[j] = (_Float16)W[k * Dm + ncol];
        }
        bfrag[t] = b;
    }

    for (int tt = 0; tt < tilesPerBlock; ++tt) {
        int tile = blockIdx.x * tilesPerBlock + tt;
        if (tile >= nRowTiles) break;
        int row0 = tile * 16;

        __syncthreads();
        {   // cooperative stage of the 16x128 f32 A tile
            const f4* xsrc = (const f4*)(X + (size_t)row0 * Dm);
            f4* sdst = (f4*)sA;
            for (int i = threadIdx.x; i < 16 * Dm / 4; i += 256) sdst[i] = xsrc[i];
        }
        __syncthreads();

        v8f acc = {};
#pragma unroll
        for (int t = 0; t < 4; ++t) {
            // A fragment 16x32 f16: row ln, K = 32t + 16h + 0..15
            const float* p = sA + ln * Dm + 32 * t + 16 * half;
            v16h a;
#pragma unroll
            for (int j = 0; j < 8; ++j) a[j]     = (_Float16)p[j];
#pragma unroll
            for (int j = 0; j < 8; ++j) a[8 + j] = (_Float16)p[8 + j];
            acc = __builtin_amdgcn_wmma_f32_16x16x32_f16(
                false, a, false, bfrag[t], (short)0, acc, false, false);
        }
        // D layout: lane column ncol, VGPR r -> row r + 8*half
        float* outp = Y + (size_t)(row0 + 8 * half) * Dm + ncol;
#pragma unroll
        for (int r = 0; r < 8; ++r) outp[(size_t)r * Dm] = acc[r];
    }
}

// ---------------------------------------------- per-graph GCN message pass
// One block per (graph, channel-half). LDS accumulator acc[1000][64] f32.
// init: acc = bias + xw*dinv^2 (self loop); edges: ds_add_f32 scatter;
// MODE 0: h = relu(acc) -> global.  MODE 1: pool = (1/K) * sum_n relu(acc).
template <int MODE>
__global__ void __launch_bounds__(256)
conv_scatter_kernel(const float* __restrict__ xw, const float* __restrict__ dinv,
                    const int* __restrict__ srcI, const int* __restrict__ dstI,
                    const float* __restrict__ bias, float* __restrict__ out,
                    float invK) {
    extern __shared__ float smem[];
    float* acc   = smem;                 // NPG*64
    float* sdinv = smem + NPG * 64;      // NPG
    float* red   = sdinv + NPG;          // 256 (MODE 1)

    const int g  = blockIdx.x >> 1;
    const int hf = blockIdx.x & 1;
    const int tid = threadIdx.x;
    const int nodeBase = g * NPG;

    for (int i = tid; i < NPG; i += 256) sdinv[i] = dinv[nodeBase + i];
    __syncthreads();

    // init with self-loop term + bias (vectorized by float4)
    for (int i = tid; i < NPG * 16; i += 256) {
        int n  = i >> 4;
        int c4 = (i & 15) << 2;
        float dv = sdinv[n];
        float w2 = dv * dv;
        f4 xv = *(const f4*)(xw + (size_t)(nodeBase + n) * Dm + hf * 64 + c4);
        f4 bv = *(const f4*)(bias + hf * 64 + c4);
        f4 r; r.x = bv.x + xv.x * w2; r.y = bv.y + xv.y * w2;
              r.z = bv.z + xv.z * w2; r.w = bv.w + xv.w * w2;
        *(f4*)&acc[n * 64 + c4] = r;
    }
    __syncthreads();

    // edge scatter into LDS (graph-local edges, graph-sorted layout)
    const int* s = srcI + (size_t)g * EPG;
    const int* d = dstI + (size_t)g * EPG;
    for (int e = tid; e < EPG; e += 256) {
        int gs = s[e], gd = d[e];
        int ls = gs - nodeBase, ld = gd - nodeBase;
        float w = sdinv[ls] * sdinv[ld];
        const float* xp = xw + (size_t)gs * Dm + hf * 64;
        float* ap = acc + ld * 64;
#pragma unroll
        for (int c = 0; c < 64; c += 4) {
            f4 v = *(const f4*)(xp + c);
            atomicAdd(ap + c + 0, v.x * w);
            atomicAdd(ap + c + 1, v.y * w);
            atomicAdd(ap + c + 2, v.z * w);
            atomicAdd(ap + c + 3, v.w * w);
        }
    }
    __syncthreads();

    if (MODE == 0) {
        for (int i = tid; i < NPG * 16; i += 256) {
            int n  = i >> 4;
            int c4 = (i & 15) << 2;
            f4 v = *(const f4*)&acc[n * 64 + c4];
            v.x = fmaxf(v.x, 0.f); v.y = fmaxf(v.y, 0.f);
            v.z = fmaxf(v.z, 0.f); v.w = fmaxf(v.w, 0.f);
            *(f4*)(out + (size_t)(nodeBase + n) * Dm + hf * 64 + c4) = v;
        }
    } else {
        // fused mean pool: pool[g][hf*64+c] = invK * sum_n relu(acc[n][c])
        int c = tid & 63;
        int part = tid >> 6;                     // 4 partial sums per channel
        float sum = 0.f;
        for (int n = part; n < NPG; n += 4) sum += fmaxf(acc[n * 64 + c], 0.f);
        red[tid] = sum;
        __syncthreads();
        if (part == 0) {
            float tot = red[c] + red[64 + c] + red[128 + c] + red[192 + c];
            out[g * Dm + hf * 64 + c] = tot * invK;
        }
    }
}

// ------------------------------------------------------------------- head
// out = relu(pool@W1 + b1) @ W2 + b2   (128x128 @ 128x128, then 128x10)
__global__ void __launch_bounds__(256)
head_kernel(const float* __restrict__ pool, const float* __restrict__ W1,
            const float* __restrict__ b1, const float* __restrict__ W2,
            const float* __restrict__ b2, float* __restrict__ out) {
    extern __shared__ float t[];   // 128*128
    for (int i = threadIdx.x; i < Bg * Dm; i += 256) {
        int bb = i >> 7, dd = i & 127;
        float s = b1[dd];
        for (int k = 0; k < Dm; ++k) s += pool[bb * Dm + k] * W1[k * Dm + dd];
        t[i] = fmaxf(s, 0.f);
    }
    __syncthreads();
    for (int i = threadIdx.x; i < Bg * Cc; i += 256) {
        int bb = i / Cc, cc = i % Cc;
        float s = b2[cc];
        for (int k = 0; k < Dm; ++k) s += t[bb * Dm + k] * W2[k * Cc + cc];
        out[i] = s;
    }
}

// ------------------------------------------------------------------ launch
extern "C" void kernel_launch(void* const* d_in, const int* in_sizes, int n_in,
                              void* d_out, int out_size, void* d_ws, size_t ws_size,
                              hipStream_t stream) {
    const float* x     = (const float*)d_in[0];
    const int*   ei    = (const int*)  d_in[1];
    const int*   srcI  = ei;
    const int*   dstI  = ei + Ee;
    const float* W_pre = (const float*)d_in[3];
    const float* b_pre = (const float*)d_in[4];
    const float* W_emb = (const float*)d_in[5];
    const float* b_emb = (const float*)d_in[6];
    // d_in[7] W_asn, d_in[8] b_asn: provably unused (softmax rows sum to 1)
    const float* W1    = (const float*)d_in[9];
    const float* b1    = (const float*)d_in[10];
    const float* W2    = (const float*)d_in[11];
    const float* b2    = (const float*)d_in[12];
    float* out = (float*)d_out;

    // workspace layout
    float* xw   = (float*)d_ws;                       // [N,128]
    float* h    = xw + (size_t)Nn * Dm;               // [N,128]
    int*   deg  = (int*)(h + (size_t)Nn * Dm);        // [N]
    float* dinv = (float*)(deg + Nn);                 // [N]
    float* pool = dinv + Nn;                          // [B,128]

    // degrees -> dinv
    zero_i32_kernel<<<(Nn + 255) / 256, 256, 0, stream>>>(deg, Nn);
    deg_count_kernel<<<(Ee + 255) / 256, 256, 0, stream>>>(dstI, Ee, deg);
    make_dinv_kernel<<<(Nn + 255) / 256, 256, 0, stream>>>(deg, dinv, Nn);

    const int nRowTiles = Nn / 16;                    // 8000
    const int tilesPerBlock = 8;
    const int gemmGrid = nRowTiles / tilesPerBlock;   // 1000
    const size_t gemmSmem = 16 * Dm * sizeof(float);  // 8KB
    const size_t scatSmem = (size_t)(NPG * 64 + NPG + 256) * sizeof(float); // ~255KB

    // conv1: h = relu(GCN(x, W_pre) + b_pre)
    gemm128_wmma_kernel<<<gemmGrid, 256, gemmSmem, stream>>>(x, W_pre, xw, nRowTiles, tilesPerBlock);
    conv_scatter_kernel<0><<<Bg * 2, 256, scatSmem, stream>>>(xw, dinv, srcI, dstI, b_pre, h, 0.f);

    // conv2 + fused mean pool: pool[b,:] = (1/K) * sum_n relu(GCN(h, W_emb) + b_emb)
    gemm128_wmma_kernel<<<gemmGrid, 256, gemmSmem, stream>>>(h, W_emb, xw, nRowTiles, tilesPerBlock);
    conv_scatter_kernel<1><<<Bg * 2, 256, scatSmem, stream>>>(xw, dinv, srcI, dstI, b_emb, pool, 1.0f / (float)Kc);

    // MLP head
    head_kernel<<<1, 256, Bg * Dm * sizeof(float), stream>>>(pool, W1, b1, W2, b2, out);
}
